// Modnet_13572096655546
// MI455X (gfx1250) — matrix operand order repaired and checked by reading, not compiled, not run on hardware
//
#include <hip/hip_runtime.h>
#include <hip/hip_bf16.h>
#include <math.h>

typedef __attribute__((ext_vector_type(16))) _Float16     v16h;
typedef __attribute__((ext_vector_type(8)))  float        v8f;
typedef __attribute__((ext_vector_type(4)))  unsigned int v4u;
typedef __attribute__((ext_vector_type(8)))  int          v8i;
typedef __attribute__((ext_vector_type(4)))  int          v4i;

#define B_N   2048
#define D_INN 16
#define Q_N   512
#define H_N   128
#define CQ    32                    // q-rows per LDS chunk (16 KB)
#define NCHUNK (Q_N / CQ)           // 16
#define CHUNK_ELEMS (CQ * H_N)      // 4096 floats

#if __has_builtin(__builtin_amdgcn_tanhf)
#define TANHF(x) __builtin_amdgcn_tanhf(x)
#else
#define TANHF(x) tanhf(x)
#endif

// ---------------------------------------------------------------------------
// Kernel 1: y[q] = sin(quad_x[q] . eq_param), plus sum_q y[q]
// ---------------------------------------------------------------------------
__global__ void __launch_bounds__(512) k_quad(const float* __restrict__ qx,
                                              const float* __restrict__ eq,
                                              float* __restrict__ y,
                                              float* __restrict__ sumy) {
    __shared__ float red[512];
    const int q = threadIdx.x;
    float v = sinf(qx[2 * q] * eq[0] + qx[2 * q + 1] * eq[1]);
    y[q]   = v;
    red[q] = v;
    __syncthreads();
    for (int s = 256; s > 0; s >>= 1) {
        if (q < s) red[q] += red[q + s];
        __syncthreads();
    }
    if (q == 0) sumy[0] = red[0];
}

// ---------------------------------------------------------------------------
// Kernel 2: Aq[q,h] = quad_x[q,0]*W1[0,h] + quad_x[q,1]*W1[1,h] + b1[h]
// ---------------------------------------------------------------------------
__global__ void __launch_bounds__(256) k_aq(const float* __restrict__ qx,
                                            const float* __restrict__ W1,
                                            const float* __restrict__ b1,
                                            float* __restrict__ A) {
    const int idx = blockIdx.x * 256 + threadIdx.x;   // 65536 total
    const int q = idx >> 7, h = idx & 127;
    A[idx] = qx[2 * q] * W1[h] + qx[2 * q + 1] * W1[H_N + h] + b1[h];
}

// ---------------------------------------------------------------------------
// Kernel 3 (WMMA): Z[b,h] = input[b,:] @ W1[2:18,h]
// M=2048, N=128, K=16 (zero-padded to 32). One v_wmma_f32_16x16x32_f16 per
// 16x16 tile; 1024 tile jobs, 8 waves/block, 128 blocks.
// Fragment layouts per CDNA5 ISA 7.12.2 (wave32 f16 A 16x32 / B 32x16).
// ---------------------------------------------------------------------------
__global__ void __launch_bounds__(256) k_z(const float* __restrict__ input,
                                           const float* __restrict__ W1,
                                           float* __restrict__ Z) {
    const int lane = threadIdx.x & 31;
    const int wave = threadIdx.x >> 5;
    const int job  = blockIdx.x * 8 + wave;   // 0..1023
    const int mt = job >> 3, nt = job & 7;
    const int mBase = mt * 16, nBase = nt * 16;
    const int kh = lane >> 4;                 // lane half (0/1)
    const int ln = lane & 15;

    // A fragment: 16x32 f16. lanes0-15: V0..3 -> K=0..7, V4..7 -> K=16..23
    //                        lanes16-31: +8.  Only K<16 carries data.
    v16h a = {};
    #pragma unroll
    for (int v = 0; v < 8; ++v) {
        #pragma unroll
        for (int hh = 0; hh < 2; ++hh) {
            const int K = 8 * kh + 2 * v + (v >= 4 ? 8 : 0) + hh;
            const float x = (K < D_INN) ? input[(mBase + ln) * D_INN + K] : 0.0f;
            a[2 * v + hh] = (_Float16)x;
        }
    }
    // B fragment: 32x16 f16. lanes0-15: V0..3 -> K=0..7, V4..7 -> K=16..23; lanes16-31: +8
    v16h bm = {};
    #pragma unroll
    for (int v = 0; v < 8; ++v) {
        #pragma unroll
        for (int hh = 0; hh < 2; ++hh) {
            const int K = 8 * kh + 2 * v + (v >= 4 ? 8 : 0) + hh;
            const float x = (K < D_INN) ? W1[(2 + K) * H_N + nBase + ln] : 0.0f;
            bm[2 * v + hh] = (_Float16)x;
        }
    }
    v8f c = {};
    c = __builtin_amdgcn_wmma_f32_16x16x32_f16(false, a, false, bm,
                                               (short)0, c, false, false);
    // C/D layout: VGPR r, lanes0-15 -> M=r, lanes16-31 -> M=r+8; N=lane&15
    #pragma unroll
    for (int r = 0; r < 8; ++r)
        Z[(mBase + r + 8 * kh) * H_N + nBase + ln] = c[r];
}

// ---------------------------------------------------------------------------
// TDM: DMA one chunk of Aq (CHUNK_ELEMS f32, contiguous) from global to LDS.
// D# per CDNA5 ISA ch.8: group0 {count=1, lds_addr, global_addr, type=2},
// group1 {data_size=4B, tensor_dim0=tile_dim0=4096, stride=4096}.
// ---------------------------------------------------------------------------
__device__ __forceinline__ void tdm_load_chunk(const float* gsrc, unsigned ldsAddr) {
    const unsigned long long ga = (unsigned long long)(uintptr_t)gsrc;
    v4u g0;
    g0[0] = 1u;                                     // count=1 (valid user D#)
    g0[1] = ldsAddr;                                // lds_addr (bytes)
    g0[2] = (unsigned)ga;                           // global_addr[31:0]
    g0[3] = (unsigned)((ga >> 32) & 0x01FFFFFFull)  // global_addr[56:32]
            | (2u << 30);                           // type=2 ("image")
    v8i g1;
    g1[0] = 2 << 16;                                // data_size=2 -> 4 bytes
    g1[1] = (int)((CHUNK_ELEMS & 0xFFFF) << 16);    // tensor_dim0[15:0]
    g1[2] = (int)((CHUNK_ELEMS >> 16) & 0xFFFF)     // tensor_dim0[31:16]
            | (1 << 16);                            // tensor_dim1 = 1
    g1[3] = (int)(CHUNK_ELEMS << 16);               // tile_dim0 = 4096
    g1[4] = 0;                                      // tile_dim1/2 unused
    g1[5] = CHUNK_ELEMS;                            // tensor_dim0_stride[31:0]
    g1[6] = 0;
    g1[7] = 0;
    v4i z4 = {};
    v8i z8 = {};
#if __clang_major__ >= 23
    __builtin_amdgcn_tensor_load_to_lds(g0, g1, z4, z4, z8, 0);
#else
    __builtin_amdgcn_tensor_load_to_lds(g0, g1, z4, z4, 0);
#endif
    (void)z8;
}

// ---------------------------------------------------------------------------
// Kernel 4 (main): out[b] = sum_h W2[h] * sum_q y[q]*tanh(Aq[q,h]+Z[b,h])
//                          + b2 * sum_q y[q]
// 8 waves/block, 1 wave per b, 4 h-channels per lane. Aq double-buffered in
// LDS via TDM + s_wait_tensorcnt; y cached in LDS.
// ---------------------------------------------------------------------------
__global__ void __launch_bounds__(256) k_main(const float* __restrict__ A_g,
                                              const float* __restrict__ Z,
                                              const float* __restrict__ y_g,
                                              const float* __restrict__ sumy,
                                              const float* __restrict__ W2,
                                              const float* __restrict__ b2,
                                              float* __restrict__ out) {
    extern __shared__ float smem[];
    float* Abuf = smem;                       // 2 * CHUNK_ELEMS floats
    float* ylds = smem + 2 * CHUNK_ELEMS;     // Q_N floats

    const int lane = threadIdx.x & 31;
    const int wave = threadIdx.x >> 5;
    const int b    = blockIdx.x * 8 + wave;

    for (int i = threadIdx.x; i < Q_N; i += 256) ylds[i] = y_g[i];

    const float4 z  = ((const float4*)(Z + b * H_N))[lane];  // Z[b, 4l..4l+3]
    const float4 w2 = ((const float4*)W2)[lane];             // W2[4l..4l+3]

    const unsigned ldsBase = (unsigned)(uintptr_t)(void*)Abuf; // LDS byte addr

    if (wave == 0) {                         // wave-level TDM issue (EXEC ignored)
        tdm_load_chunk(A_g, ldsBase);
        __builtin_amdgcn_s_wait_tensorcnt(0);
    }
    __syncthreads();

    float acc0 = 0.f, acc1 = 0.f, acc2 = 0.f, acc3 = 0.f;
    for (int c = 0; c < NCHUNK; ++c) {
        if (wave == 0 && c + 1 < NCHUNK)     // prefetch next chunk into other buffer
            tdm_load_chunk(A_g + (size_t)(c + 1) * CHUNK_ELEMS,
                           ldsBase + (unsigned)(((c + 1) & 1) * CHUNK_ELEMS * 4));
        const float* Ach = Abuf + (c & 1) * CHUNK_ELEMS;
        #pragma unroll 4
        for (int q = 0; q < CQ; ++q) {
            const float4 aq = ((const float4*)(Ach + q * H_N))[lane];
            const float  yq = ylds[c * CQ + q];
            acc0 = fmaf(yq, TANHF(aq.x + z.x), acc0);
            acc1 = fmaf(yq, TANHF(aq.y + z.y), acc1);
            acc2 = fmaf(yq, TANHF(aq.z + z.z), acc2);
            acc3 = fmaf(yq, TANHF(aq.w + z.w), acc3);
        }
        __syncthreads();                      // everyone done with buf[c&1]
        if (wave == 0) __builtin_amdgcn_s_wait_tensorcnt(0); // next chunk landed
        __syncthreads();
    }

    float p = acc0 * w2.x + acc1 * w2.y + acc2 * w2.z + acc3 * w2.w;
    #pragma unroll
    for (int off = 16; off > 0; off >>= 1)    // wave32 reduction
        p += __shfl_xor(p, off, 32);
    if (lane == 0) out[b] = p + b2[0] * sumy[0];
}

// ---------------------------------------------------------------------------
extern "C" void kernel_launch(void* const* d_in, const int* in_sizes, int n_in,
                              void* d_out, int out_size, void* d_ws, size_t ws_size,
                              hipStream_t stream) {
    const float* input = (const float*)d_in[0];   // [2048,16]
    const float* eq    = (const float*)d_in[1];   // [2,1]
    const float* qx    = (const float*)d_in[2];   // [512,2]
    const float* W1    = (const float*)d_in[3];   // [18,128]
    const float* b1    = (const float*)d_in[4];   // [128]
    const float* W2    = (const float*)d_in[5];   // [128,1]
    const float* b2    = (const float*)d_in[6];   // [1]
    float* out = (float*)d_out;                   // [2048,1]

    float* wsf  = (float*)d_ws;
    float* A    = wsf;                        // 512*128   = 65536 floats
    float* Z    = wsf + 65536;                // 2048*128  = 262144 floats
    float* y    = wsf + 65536 + 262144;       // 512 floats
    float* sumy = y + Q_N;                    // 1 float

    k_quad<<<1, 512, 0, stream>>>(qx, eq, y, sumy);
    k_aq  <<<256, 256, 0, stream>>>(qx, W1, b1, A);
    k_z   <<<128, 256, 0, stream>>>(input, W1, Z);

    const size_t sh = (2 * CHUNK_ELEMS + Q_N) * sizeof(float);  // 34 KB
    k_main<<<256, 256, sh, stream>>>(A, Z, y, sumy, W2, b2, out);
    (void)in_sizes; (void)n_in; (void)out_size; (void)ws_size;
}
